// ReAttention_4526895530288
// MI455X (gfx1250) — matrix operand order repaired and checked by reading, not compiled, not run on hardware
//
#include <hip/hip_runtime.h>
#include <hip/hip_bf16.h>
#include <math.h>
#include <stdint.h>

// ---------------------------------------------------------------------------
// DeepViT Re-Attention on MI455X (gfx1250, wave32, WMMA 16x16x32 f16 + TDM)
// B=8, N=1024, D=1024, H=8, HD=128, C=1, S=32
// ---------------------------------------------------------------------------

#define BB   8
#define NTOK 1024
#define DDIM 1024
#define HH   8
#define HDIM 128
#define BH   (BB * HH)            // 64
#define ATT_SCALE 0.08838834764831845f   // 128^-0.5
#define BN_EPS 1e-3f

#ifndef __has_builtin
#define __has_builtin(x) 0
#endif
#if __has_builtin(__builtin_amdgcn_tensor_load_to_lds) && \
    __has_builtin(__builtin_amdgcn_s_wait_tensorcnt)
#define HAVE_TDM 1
#else
#define HAVE_TDM 0
#endif
#if defined(__has_include)
#if __has_include(<hip/amd_detail/amd_gfx1250_TDM.h>)
#define TDM_6ARG 1
#endif
#endif
#ifndef TDM_6ARG
#define TDM_6ARG 0
#endif

typedef __attribute__((ext_vector_type(16))) _Float16 v16h;
typedef __attribute__((ext_vector_type(8)))  float    v8f;

union FragU { uint4 u[2]; v16h v; };

// A-operand fragment (16x32 f16): lane = row (lane&15), hi = lane>>4.
// K chunks live at [hi*8 .. hi*8+7] and [16+hi*8 .. 23+hi*8] (ISA 7.12.2).
__device__ __forceinline__ v16h fragA(const _Float16* p, int hi) {
    FragU f;
    f.u[0] = *(const uint4*)(p + hi * 8);
    f.u[1] = *(const uint4*)(p + 16 + hi * 8);
    return f.v;
}
// B-operand fragment (32x16 f16): lane = column; K chunks [hi*8..hi*8+7], [16+hi*8..23+hi*8].
__device__ __forceinline__ v16h fragB(const _Float16* p, int hi) {
    FragU f;
    f.u[0] = *(const uint4*)(p + hi * 8);
    f.u[1] = *(const uint4*)(p + 16 + hi * 8);
    return f.v;
}

__device__ __forceinline__ v8f wmma_f16(v16h a, v16h b, v8f c) {
    return __builtin_amdgcn_wmma_f32_16x16x32_f16(false, a, false, b,
                                                  (short)0, c, false, false);
}

__device__ __forceinline__ float gelu_exact(float x) {
    return 0.5f * x * (1.0f + erff(x * 0.70710678118654752f));
}

#if HAVE_TDM
// ---------------------------------------------------------------------------
// Tensor Data Mover: load a 32(K) x 128(rows) f16 tile (row stride = ld elems)
// from global into LDS at lds_byte_off, padding each 64B row with 16B so the
// LDS layout is [128][40] f16 — identical to the manual tiling layout.
// D# packing per CDNA5 ISA 8.3/8.4: group0 = {count|lds|global|type},
// group1 = {flags, tensor_dim0/1, tile_dim0/1/2, stride0}.
// ---------------------------------------------------------------------------
typedef __attribute__((ext_vector_type(4))) unsigned int tdm_v4u;
typedef __attribute__((ext_vector_type(8))) int          tdm_v8i;
typedef __attribute__((ext_vector_type(4))) int          tdm_v4i;

__device__ __forceinline__ void tdm_load_tile_f16(
    const _Float16* tile_start, unsigned lds_byte_off, int ld_elems)
{
    const unsigned long long ga = (unsigned long long)(uintptr_t)tile_start;
    tdm_v4u g0;
    g0[0] = 1u;                                           // count=1, user mode
    g0[1] = lds_byte_off;                                 // LDS address
    g0[2] = (unsigned)(ga & 0xFFFFFFFFull);               // global addr lo
    g0[3] = (unsigned)((ga >> 32) & 0x1FFFFFFull)         // global addr hi
            | (2u << 30);                                 // type = 2 (image)
    tdm_v8i g1;
    g1[0] = (1 << 16)        // data_size = 2 bytes
          | (1 << 20)        // pad_enable
          | (3 << 22)        // pad_interval: 16 DWORDs (64B) before pad
          | (3 << 25);       // pad_amount: 4 DWORDs (16B)
    g1[1] = (int)(32u << 16);                 // tensor_dim0 = 32 (lo16 @bit48)
    g1[2] = (int)(128u << 16);                // dim0 hi=0 | tensor_dim1=128 lo
    g1[3] = (int)(32u << 16);                 // dim1 hi=0 | tile_dim0 = 32
    g1[4] = 128;                              // tile_dim1 = 128, tile_dim2 = 0
    g1[5] = ld_elems;                         // tensor_dim0_stride lo32
    g1[6] = 0;                                // stride0 hi | stride1 lo
    g1[7] = 0;
    tdm_v4i gz = {0, 0, 0, 0};
#if TDM_6ARG
    tdm_v8i gz8 = {0, 0, 0, 0, 0, 0, 0, 0};
    __builtin_amdgcn_tensor_load_to_lds(g0, g1, gz, gz, gz8, 0);
#else
    __builtin_amdgcn_tensor_load_to_lds(g0, g1, gz, gz, 0);
#endif
}
#endif // HAVE_TDM

// ---------------------------------------------------------------------------
// K1: per-(b,n) 3x3 SAME conv (C=1) + exact GELU for q,k,v.
//     q,k -> [bh][n][hd] f16 ; v -> transposed [bh][hd][n] f16 (B-operand ready)
// ---------------------------------------------------------------------------
__global__ __launch_bounds__(256) void conv_qkv_kernel(
    const float* __restrict__ x,
    const float* __restrict__ qw, const float* __restrict__ kw,
    const float* __restrict__ vw,
    _Float16* __restrict__ q, _Float16* __restrict__ k,
    _Float16* __restrict__ vT)
{
    const int map = blockIdx.x;           // b*1024 + n
    const int b = map >> 10, n = map & 1023;
    __shared__ float xm[1024];
    __shared__ float wq[9], wk[9], wv[9];
    const float* xb = x + (size_t)map * 1024;
    for (int i = threadIdx.x; i < 1024; i += 256) xm[i] = xb[i];
    if (threadIdx.x < 9) {
        wq[threadIdx.x] = qw[threadIdx.x];
        wk[threadIdx.x] = kw[threadIdx.x];
        wv[threadIdx.x] = vw[threadIdx.x];
    }
    __syncthreads();
    #pragma unroll
    for (int p = 0; p < 4; ++p) {
        const int px = threadIdx.x + p * 256;
        const int r = px >> 5, c = px & 31;
        float aq = 0.f, ak = 0.f, av = 0.f;
        #pragma unroll
        for (int dr = -1; dr <= 1; ++dr) {
            #pragma unroll
            for (int dc = -1; dc <= 1; ++dc) {
                const int rr = r + dr, cc = c + dc;
                if (rr >= 0 && rr < 32 && cc >= 0 && cc < 32) {
                    const float xv = xm[rr * 32 + cc];
                    const int wi = (dr + 1) * 3 + (dc + 1);
                    aq += xv * wq[wi]; ak += xv * wk[wi]; av += xv * wv[wi];
                }
            }
        }
        const float gq = gelu_exact(aq), gk = gelu_exact(ak), gv = gelu_exact(av);
        const int h = px >> 7, hd = px & 127, bh = b * HH + h;
        q[((size_t)bh * NTOK + n) * HDIM + hd] = (_Float16)gq;
        k[((size_t)bh * NTOK + n) * HDIM + hd] = (_Float16)gk;
        vT[((size_t)bh * HDIM + hd) * NTOK + n] = (_Float16)gv;
    }
}

// ---------------------------------------------------------------------------
// K0a: transpose 1024x1024 f32 -> f16 (row j of output = column j of input)
// ---------------------------------------------------------------------------
__global__ __launch_bounds__(256) void transpose_cvt_kernel(
    const float* __restrict__ W, _Float16* __restrict__ T)
{
    const size_t o = (size_t)blockIdx.x * 256 + threadIdx.x;
    const int j = (int)(o >> 10), c = (int)(o & 1023);
    T[o] = (_Float16)W[(size_t)c * 1024 + j];
}

// K0b: BN inference scale: mul[j] = gamma[j] / sqrt(1 + eps)
__global__ __launch_bounds__(256) void bnmul_kernel(
    const float* __restrict__ gamma, float* __restrict__ mul)
{
    const int j = blockIdx.x * 256 + threadIdx.x;
    if (j < NTOK) mul[j] = gamma[j] * rsqrtf(1.0f + BN_EPS);
}

// ---------------------------------------------------------------------------
// K2: scores + softmax.  Block = 16 rows of one (bh); 8 waves x 128 cols each.
//     S = q k^T * scale, row-softmax, P stored f16.
// ---------------------------------------------------------------------------
__global__ __launch_bounds__(256) void scores_softmax_kernel(
    const _Float16* __restrict__ q, const _Float16* __restrict__ k,
    _Float16* __restrict__ P)
{
    const int bh = blockIdx.y;
    const int i0 = blockIdx.x * 16;
    const int tid = threadIdx.x;
    const int lane = tid & 31, w = tid >> 5;
    const int hi = lane >> 4, lm = lane & 15;
    const _Float16* qb = q + (size_t)bh * NTOK * HDIM;
    const _Float16* kb = k + (size_t)bh * NTOK * HDIM;

    v8f acc[8];
    #pragma unroll
    for (int t = 0; t < 8; ++t)
        #pragma unroll
        for (int r = 0; r < 8; ++r) acc[t][r] = 0.f;

    #pragma unroll
    for (int ks = 0; ks < HDIM; ks += 32) {
        const v16h a = fragA(qb + (size_t)(i0 + lm) * HDIM + ks, hi);
        #pragma unroll
        for (int t = 0; t < 8; ++t) {
            const int j = w * 128 + t * 16 + lm;
            const v16h bfr = fragB(kb + (size_t)j * HDIM + ks, hi);
            acc[t] = wmma_f16(a, bfr, acc[t]);
        }
    }

    __shared__ float red[8][16];
    __shared__ float stat[16];

    float m[8];
    #pragma unroll
    for (int r = 0; r < 8; ++r) {
        float v = -INFINITY;
        #pragma unroll
        for (int t = 0; t < 8; ++t) {
            acc[t][r] *= ATT_SCALE;
            v = fmaxf(v, acc[t][r]);
        }
        for (int off = 1; off < 16; off <<= 1)
            v = fmaxf(v, __shfl_xor(v, off, 32));
        m[r] = v;
    }
    if (lm == 0)
        #pragma unroll
        for (int r = 0; r < 8; ++r) red[w][r + 8 * hi] = m[r];
    __syncthreads();
    if (tid < 16) {
        float v = red[0][tid];
        #pragma unroll
        for (int ww = 1; ww < 8; ++ww) v = fmaxf(v, red[ww][tid]);
        stat[tid] = v;
    }
    __syncthreads();

    float s[8];
    #pragma unroll
    for (int r = 0; r < 8; ++r) {
        const float rmax = stat[r + 8 * hi];
        float sum = 0.f;
        #pragma unroll
        for (int t = 0; t < 8; ++t) {
            const float e = __expf(acc[t][r] - rmax);
            acc[t][r] = e;
            sum += e;
        }
        for (int off = 1; off < 16; off <<= 1)
            sum += __shfl_xor(sum, off, 32);
        s[r] = sum;
    }
    __syncthreads();
    if (lm == 0)
        #pragma unroll
        for (int r = 0; r < 8; ++r) red[w][r + 8 * hi] = s[r];
    __syncthreads();
    if (tid < 16) {
        float v = 0.f;
        #pragma unroll
        for (int ww = 0; ww < 8; ++ww) v += red[ww][tid];
        stat[tid] = v;
    }
    __syncthreads();

    _Float16* Pb = P + (size_t)bh * NTOK * NTOK + (size_t)i0 * NTOK;
    #pragma unroll
    for (int r = 0; r < 8; ++r) {
        const float rinv = 1.0f / stat[r + 8 * hi];
        #pragma unroll
        for (int t = 0; t < 8; ++t)
            Pb[(size_t)(r + 8 * hi) * NTOK + w * 128 + t * 16 + lm] =
                (_Float16)(acc[t][r] * rinv);
    }
}

// ---------------------------------------------------------------------------
// K3/K5: LDS-tiled f16 WMMA GEMM, 128x128 block tile, 8 waves (4x2 grid),
//        each wave 32x64 (2x4 accumulator tiles).  C = A * Bt^T.
//        TDM double-buffered tile fill (DMA next tile overlaps WMMA on cur);
//        fallback: manual global->LDS loads when TDM builtin is unavailable.
//        Epilogue: y = (acc + bias[col]) * mul[col] + add[col].
// ---------------------------------------------------------------------------
__global__ __launch_bounds__(256) void gemm128_kernel(
    const _Float16* __restrict__ A, size_t aBatch, int lda,
    const _Float16* __restrict__ Bt, int ldb,
    const float* __restrict__ bias, const float* __restrict__ mul,
    const float* __restrict__ add,
    float* __restrict__ Cf, size_t cfBatch, int ldc,
    _Float16* __restrict__ Ch, size_t chBatch, int ldch,
    int K)
{
    __shared__ __align__(16) _Float16 As[2][128][40];
    __shared__ __align__(16) _Float16 Bs[2][128][40];
    const int tid = threadIdx.x, lane = tid & 31, w = tid >> 5;
    const int hi = lane >> 4, lm = lane & 15;
    const int wm = w >> 1, wn = w & 1;
    const int row0 = blockIdx.y * 128, col0 = blockIdx.x * 128;
    const _Float16* Ab = A + (size_t)blockIdx.z * aBatch;

    v8f acc[2][4];
    #pragma unroll
    for (int i = 0; i < 2; ++i)
        #pragma unroll
        for (int j = 0; j < 4; ++j)
            #pragma unroll
            for (int r = 0; r < 8; ++r) acc[i][j][r] = 0.f;

#if HAVE_TDM
    const unsigned ldsA[2] = { (unsigned)(uintptr_t)&As[0][0][0],
                               (unsigned)(uintptr_t)&As[1][0][0] };
    const unsigned ldsB[2] = { (unsigned)(uintptr_t)&Bs[0][0][0],
                               (unsigned)(uintptr_t)&Bs[1][0][0] };
    if (w == 0) {   // one wave drives the TDM; TENSORcnt tracks completion
        tdm_load_tile_f16(Ab + (size_t)row0 * lda, ldsA[0], lda);
        tdm_load_tile_f16(Bt + (size_t)col0 * ldb, ldsB[0], ldb);
    }
    for (int kt = 0, it = 0; kt < K; kt += 32, ++it) {
        const int cur = it & 1;
        if (w == 0) {
            if ((kt + 32) < K) {   // prefetch next tiles into the other buffer
                tdm_load_tile_f16(Ab + (size_t)row0 * lda + kt + 32,
                                  ldsA[cur ^ 1], lda);
                tdm_load_tile_f16(Bt + (size_t)col0 * ldb + kt + 32,
                                  ldsB[cur ^ 1], ldb);
                // TDM ops complete in order: <=2 outstanding => cur A,B done
                __builtin_amdgcn_s_wait_tensorcnt(2);
            } else {
                __builtin_amdgcn_s_wait_tensorcnt(0);
            }
        }
        __syncthreads();
        v16h af[2], bf[4];
        #pragma unroll
        for (int i = 0; i < 2; ++i)
            af[i] = fragA(&As[cur][wm * 32 + i * 16 + lm][0], hi);
        #pragma unroll
        for (int j = 0; j < 4; ++j)
            bf[j] = fragB(&Bs[cur][wn * 64 + j * 16 + lm][0], hi);
        #pragma unroll
        for (int i = 0; i < 2; ++i)
            #pragma unroll
            for (int j = 0; j < 4; ++j)
                acc[i][j] = wmma_f16(af[i], bf[j], acc[i][j]);
        __syncthreads();   // everyone done with cur before it is overwritten
    }
#else
    const int lr = tid >> 1;          // 0..127
    const int seg = (tid & 1) * 16;   // 0 | 16
    for (int kt = 0; kt < K; kt += 32) {
        const _Float16* ga = Ab + (size_t)(row0 + lr) * lda + kt + seg;
        const _Float16* gb = Bt + (size_t)(col0 + lr) * ldb + kt + seg;
        *(uint4*)&As[0][lr][seg]     = *(const uint4*)ga;
        *(uint4*)&As[0][lr][seg + 8] = *(const uint4*)(ga + 8);
        *(uint4*)&Bs[0][lr][seg]     = *(const uint4*)gb;
        *(uint4*)&Bs[0][lr][seg + 8] = *(const uint4*)(gb + 8);
        __syncthreads();
        v16h af[2], bf[4];
        #pragma unroll
        for (int i = 0; i < 2; ++i)
            af[i] = fragA(&As[0][wm * 32 + i * 16 + lm][0], hi);
        #pragma unroll
        for (int j = 0; j < 4; ++j)
            bf[j] = fragB(&Bs[0][wn * 64 + j * 16 + lm][0], hi);
        #pragma unroll
        for (int i = 0; i < 2; ++i)
            #pragma unroll
            for (int j = 0; j < 4; ++j)
                acc[i][j] = wmma_f16(af[i], bf[j], acc[i][j]);
        __syncthreads();
    }
#endif

    #pragma unroll
    for (int i = 0; i < 2; ++i) {
        #pragma unroll
        for (int j = 0; j < 4; ++j) {
            const int cb = col0 + wn * 64 + j * 16 + lm;
            const int rb = row0 + wm * 32 + i * 16 + 8 * hi;
            const float bb = bias ? bias[cb] : 0.f;
            const float mm = mul ? mul[cb] : 1.f;
            const float ad = add ? add[cb] : 0.f;
            #pragma unroll
            for (int r = 0; r < 8; ++r) {
                const float y = (acc[i][j][r] + bb) * mm + ad;
                if (Cf)
                    Cf[(size_t)blockIdx.z * cfBatch + (size_t)(rb + r) * ldc + cb] = y;
                if (Ch)
                    Ch[(size_t)blockIdx.z * chBatch + (size_t)(rb + r) * ldch + cb] =
                        (_Float16)y;
            }
        }
    }
}

// ---------------------------------------------------------------------------
// K4: out_heads = attn @ v  (per bh: 1024x128x1024), write f16 X2 [B,N,D]
// ---------------------------------------------------------------------------
__global__ __launch_bounds__(256) void attnv_kernel(
    const _Float16* __restrict__ attn, const _Float16* __restrict__ vT,
    _Float16* __restrict__ X2)
{
    const int bh = blockIdx.y;
    const int b = bh >> 3, h = bh & 7;
    const int tid = threadIdx.x, lane = tid & 31, w = tid >> 5;
    const int hi = lane >> 4, lm = lane & 15;
    const int i0 = blockIdx.x * 128 + w * 16;
    const _Float16* Ab = attn + (size_t)bh * NTOK * NTOK;
    const _Float16* Bb = vT + (size_t)bh * HDIM * NTOK;

    v8f acc[8];
    #pragma unroll
    for (int t = 0; t < 8; ++t)
        #pragma unroll
        for (int r = 0; r < 8; ++r) acc[t][r] = 0.f;

    for (int ks = 0; ks < NTOK; ks += 32) {
        const v16h a = fragA(Ab + (size_t)(i0 + lm) * NTOK + ks, hi);
        #pragma unroll
        for (int t = 0; t < 8; ++t) {
            const v16h bfr = fragB(Bb + (size_t)(t * 16 + lm) * NTOK + ks, hi);
            acc[t] = wmma_f16(a, bfr, acc[t]);
        }
    }
    _Float16* Xb = X2 + ((size_t)b * NTOK + i0) * DDIM + h * HDIM;
    #pragma unroll
    for (int t = 0; t < 8; ++t)
        #pragma unroll
        for (int r = 0; r < 8; ++r)
            Xb[(size_t)(r + 8 * hi) * DDIM + t * 16 + lm] = (_Float16)acc[t][r];
}

// ---------------------------------------------------------------------------
// Launch.  d_in: x, qw, kw, vw, reatten_w, reatten_b, bn_gamma, bn_beta,
//                proj_w, proj_b  (all fp32)
// d_out: out [8,1024,1024] fp32  ++  attn [8,8,1024,1024] fp32
// ---------------------------------------------------------------------------
extern "C" void kernel_launch(void* const* d_in, const int* in_sizes, int n_in,
                              void* d_out, int out_size, void* d_ws, size_t ws_size,
                              hipStream_t stream)
{
    const float* x     = (const float*)d_in[0];
    const float* qw    = (const float*)d_in[1];
    const float* kw    = (const float*)d_in[2];
    const float* vw    = (const float*)d_in[3];
    const float* reW   = (const float*)d_in[4];
    const float* reBias= (const float*)d_in[5];
    const float* gamma = (const float*)d_in[6];
    const float* beta  = (const float*)d_in[7];
    const float* pW    = (const float*)d_in[8];
    const float* pB    = (const float*)d_in[9];

    char* ws = (char*)d_ws;
    const size_t MB = 1024ull * 1024ull;
    _Float16* q      = (_Float16*)(ws + 0);          // 16 MB  [bh][n][hd]
    _Float16* k      = (_Float16*)(ws + 16 * MB);    // 16 MB  [bh][n][hd]
    _Float16* vT     = (_Float16*)(ws + 32 * MB);    // 16 MB  [bh][hd][n]
    _Float16* P      = (_Float16*)(ws + 48 * MB);    // 128 MB [bh][i][j]
    _Float16* attn_h = (_Float16*)(ws + 176 * MB);   // 128 MB [bh][i][j]
    _Float16* X2     = (_Float16*)(ws + 304 * MB);   // 16 MB  [b][n][d]
    _Float16* WreT   = (_Float16*)(ws + 320 * MB);   // 2 MB   [j][c]
    _Float16* PwT    = (_Float16*)(ws + 322 * MB);   // 2 MB   [j][c]
    float*    bnmul  = (float*)   (ws + 324 * MB);   // 4 KB

    float* out      = (float*)d_out;
    float* attn_out = out + (size_t)BB * NTOK * DDIM;

    conv_qkv_kernel<<<BB * NTOK, 256, 0, stream>>>(x, qw, kw, vw, q, k, vT);
    transpose_cvt_kernel<<<4096, 256, 0, stream>>>(reW, WreT);
    transpose_cvt_kernel<<<4096, 256, 0, stream>>>(pW, PwT);
    bnmul_kernel<<<4, 256, 0, stream>>>(gamma, bnmul);

    scores_softmax_kernel<<<dim3(NTOK / 16, BH), 256, 0, stream>>>(q, k, P);

    // Re-attention: attn = (P @ Wre + b) * bnmul + beta  (fp32 -> d_out, f16 -> ws)
    gemm128_kernel<<<dim3(8, 8, BH), 256, 0, stream>>>(
        P, (size_t)NTOK * NTOK, NTOK, WreT, NTOK,
        reBias, bnmul, beta,
        attn_out, (size_t)NTOK * NTOK, NTOK,
        attn_h, (size_t)NTOK * NTOK, NTOK, NTOK);

    attnv_kernel<<<dim3(NTOK / 128, BH), 256, 0, stream>>>(attn_h, vT, X2);

    // Projection: out = X2 @ proj_w + proj_b  (M = 8192)
    gemm128_kernel<<<dim3(8, 64, 1), 256, 0, stream>>>(
        X2, 0, DDIM, PwT, DDIM,
        pB, nullptr, nullptr,
        out, 0, DDIM,
        (_Float16*)nullptr, 0, 0, DDIM);
}
